// TransformerBlock_67284957659468
// MI455X (gfx1250) — compile-verified, matrix operand order checked by reading, not run on hardware
//
#include <hip/hip_runtime.h>
#include <hip/hip_bf16.h>

// ---------------------------------------------------------------------------
// Types for CDNA5 WMMA (wave32)
// ---------------------------------------------------------------------------
typedef __attribute__((ext_vector_type(16))) __bf16 v16bf;
typedef __attribute__((ext_vector_type(8)))  __bf16 v8bf;
typedef __attribute__((ext_vector_type(8)))  float  v8f;
typedef __attribute__((ext_vector_type(4)))  int    v4i;

#define DIM    768
#define HEADS  12
#define HD     64
#define CLSK   20
#define HIDDEN 3072
#define BATCH  16
#define SEQ    596
#define BN     (BATCH * SEQ)      // 9536
#define NPADS  608                // seq padded to multiple of 32
#define LDP    616                // padded LDS row stride for attention (halves/floats)

#define BK     64                 // GEMM K-tile
#define LDT    72                 // padded LDS tile stride in halves (144B, conflict-free)

// gfx1250 async global->LDS copy path (tracked by ASYNCcnt)
#if defined(__HIP_DEVICE_COMPILE__) && \
    __has_builtin(__builtin_amdgcn_global_load_async_to_lds_b128) && \
    __has_builtin(__builtin_amdgcn_s_wait_asynccnt)
#define USE_ASYNC_LDS 1
#else
#define USE_ASYNC_LDS 0
#endif

#if USE_ASYNC_LDS
#define ASYNC_COPY_B128(gptr, lptr)                                          \
  __builtin_amdgcn_global_load_async_to_lds_b128(                            \
      (__attribute__((address_space(1))) v4i*)(gptr),                        \
      (__attribute__((address_space(3))) v4i*)(lptr), 0, 0)
#endif

__device__ __forceinline__ v8f wmma_bf16(v16bf a, v16bf b, v8f c) {
  // (neg_a, A, neg_b, B, c_mod, C, reuse_a, reuse_b)
  return __builtin_amdgcn_wmma_f32_16x16x32_bf16(false, a, false, b, (short)0, c,
                                                 false, false);
}

// Load a 16x32 (A) or 32x16 (B) bf16 fragment, row-major source, ld = row stride.
// ISA layout: lane L holds row/col L&15; lanes 0-15 hold K 0-7 & 16-23,
// lanes 16-31 hold K 8-15 & 24-31  -> two 16-byte loads per lane.
__device__ __forceinline__ v16bf load_frag(const __bf16* __restrict__ base,
                                           int ld, int k0) {
  const int lane = threadIdx.x & 31;
  const int r    = lane & 15;
  const int kg   = (lane >> 4) << 3;        // 0 or 8
  const __bf16* p = base + r * ld + k0 + kg;
  union { v16bf v; v8bf h[2]; } u;
  u.h[0] = *(const v8bf*)(p);
  u.h[1] = *(const v8bf*)(p + 16);
  return u.v;
}

// ---------------------------------------------------------------------------
// fp32 -> bf16 conversion (weights)
// ---------------------------------------------------------------------------
__global__ __launch_bounds__(256) void cvt_bf16_kernel(const float* __restrict__ in,
                                                       __bf16* __restrict__ out,
                                                       int n) {
  int i = blockIdx.x * 256 + threadIdx.x;
  if (i < n) out[i] = (__bf16)in[i];
}

// ---------------------------------------------------------------------------
// LayerNorm (row = one token), fp32 in -> bf16 out
// ---------------------------------------------------------------------------
__global__ __launch_bounds__(256) void ln_kernel(const float* __restrict__ x,
                                                 const float* __restrict__ g,
                                                 const float* __restrict__ b,
                                                 __bf16* __restrict__ out) {
  const int row = blockIdx.x;
  const float* xr = x + (size_t)row * DIM;
  float v[3], s = 0.f, s2 = 0.f;
#pragma unroll
  for (int i = 0; i < 3; ++i) {
    v[i] = xr[threadIdx.x + 256 * i];
    s  += v[i];
    s2 += v[i] * v[i];
  }
  __shared__ float rs[256], rq[256];
  rs[threadIdx.x] = s; rq[threadIdx.x] = s2;
  __syncthreads();
  for (int st = 128; st > 0; st >>= 1) {
    if (threadIdx.x < st) {
      rs[threadIdx.x] += rs[threadIdx.x + st];
      rq[threadIdx.x] += rq[threadIdx.x + st];
    }
    __syncthreads();
  }
  const float mean = rs[0] * (1.f / DIM);
  const float var  = rq[0] * (1.f / DIM) - mean * mean;
  const float inv  = rsqrtf(var + 1e-5f);
#pragma unroll
  for (int i = 0; i < 3; ++i) {
    int c = threadIdx.x + 256 * i;
    out[(size_t)row * DIM + c] = (__bf16)((v[i] - mean) * inv * g[c] + b[c]);
  }
}

// ---------------------------------------------------------------------------
// WMMA GEMM with async double-buffered LDS staging:
//   C[M,N] = act(A[M,K] @ W[N,K]^T + bias) (+resid)
// block = 128 threads (4 waves), tile 64(M) x 128(N), BK = 64.
// Waves: 2(M) x 2(N); wave tile 32x64 (2x4 accumulators).
// ---------------------------------------------------------------------------
__global__ __launch_bounds__(128) void gemm_wmma_kernel(
    const __bf16* __restrict__ A, const __bf16* __restrict__ W,
    const float* __restrict__ bias, const float* __restrict__ resid,
    float* __restrict__ outf, __bf16* __restrict__ outb,
    int M, int N, int K, int gelu) {
  __shared__ __attribute__((aligned(16))) __bf16 As[2][64 * LDT];
  __shared__ __attribute__((aligned(16))) __bf16 Bs[2][128 * LDT];

  const int tid  = threadIdx.x;
  const int lane = tid & 31;
  const int wid  = tid >> 5;
  const int row0 = blockIdx.y * 64;
  const int col0 = blockIdx.x * 128;
  const int wrow = (wid & 1) * 32;     // wave row within block tile
  const int wcol = (wid >> 1) * 64;    // wave col within block tile

  // Stage one BK-slice of A (64xBK) and W (128xBK) into LDS buffer `buf`.
  // Coalesced: consecutive threads cover consecutive 16B segments of a row.
  auto stage = [&](int buf, int k0) {
#pragma unroll
    for (int i = 0; i < 4; ++i) {                 // A: 64*64/8 = 512 chunks
      const int c = tid + i * 128;
      const int r = c >> 3, s = (c & 7) * 8;
      const __bf16* g = A + (size_t)(row0 + r) * K + k0 + s;
      __bf16* l = &As[buf][r * LDT + s];
#if USE_ASYNC_LDS
      ASYNC_COPY_B128(g, l);
#else
      *(v8bf*)l = *(const v8bf*)g;
#endif
    }
#pragma unroll
    for (int i = 0; i < 8; ++i) {                 // B: 128*64/8 = 1024 chunks
      const int c = tid + i * 128;
      const int r = c >> 3, s = (c & 7) * 8;
      const __bf16* g = W + (size_t)(col0 + r) * K + k0 + s;
      __bf16* l = &Bs[buf][r * LDT + s];
#if USE_ASYNC_LDS
      ASYNC_COPY_B128(g, l);
#else
      *(v8bf*)l = *(const v8bf*)g;
#endif
    }
  };

  v8f acc[2][4] = {};
  const int nk = K / BK;
  stage(0, 0);

  for (int kt = 0; kt < nk; ++kt) {
    const int buf = kt & 1;
    if (kt + 1 < nk) {
      stage(buf ^ 1, (kt + 1) * BK);
#if USE_ASYNC_LDS
      __builtin_amdgcn_s_wait_asynccnt(12);   // 12 in flight for next tile; this tile done
#endif
    } else {
#if USE_ASYNC_LDS
      __builtin_amdgcn_s_wait_asynccnt(0);
#endif
    }
    __syncthreads();

#pragma unroll
    for (int ks = 0; ks < BK; ks += 32) {
      // Preload all fragments for this K-step into distinct registers so the
      // ds_loads can run ahead of (and overlap with) the 8 XDL WMMA ops.
      v16bf a0 = load_frag(&As[buf][(wrow)      * LDT], LDT, ks);
      v16bf a1 = load_frag(&As[buf][(wrow + 16) * LDT], LDT, ks);
      v16bf bfr[4];
#pragma unroll
      for (int j = 0; j < 4; ++j)
        bfr[j] = load_frag(&Bs[buf][(wcol + j * 16) * LDT], LDT, ks);
#pragma unroll
      for (int j = 0; j < 4; ++j) {
        acc[0][j] = wmma_bf16(a0, bfr[j], acc[0][j]);
        acc[1][j] = wmma_bf16(a1, bfr[j], acc[1][j]);
      }
    }
    __syncthreads();
  }

  const int nlo = lane & 15;
  const int mhi = (lane >> 4) << 3;   // 0 or 8
#pragma unroll
  for (int i = 0; i < 2; ++i) {
#pragma unroll
    for (int j = 0; j < 4; ++j) {
      const int n = col0 + wcol + j * 16 + nlo;
      const float bval = bias[n];
#pragma unroll
      for (int r = 0; r < 8; ++r) {
        const int m = row0 + wrow + i * 16 + mhi + r;
        float val = acc[i][j][r] + bval;
        if (gelu) val = 0.5f * val * (1.0f + erff(val * 0.70710678118654752f));
        const size_t idx = (size_t)m * N + n;
        if (resid) val += resid[idx];
        if (outf) outf[idx] = val;
        if (outb) outb[idx] = (__bf16)val;
      }
    }
  }
}

// ---------------------------------------------------------------------------
// Repack qkv[BN,2304] (bf16) into padded per-head Q,K [bh,608,64] and V^T [bh,64,608]
// ---------------------------------------------------------------------------
__global__ __launch_bounds__(256) void repack_kernel(const __bf16* __restrict__ qkv,
                                                     __bf16* __restrict__ qp,
                                                     __bf16* __restrict__ kp,
                                                     __bf16* __restrict__ vt) {
  const int idx = blockIdx.x * 256 + threadIdx.x;       // over B*H*608*64
  if (idx >= BATCH * HEADS * NPADS * HD) return;
  const int d  = idx & (HD - 1);
  const int n  = (idx >> 6) % NPADS;
  const int bh = idx / (NPADS * HD);
  const int h  = bh % HEADS;
  const int b  = bh / HEADS;
  __bf16 qv = (__bf16)0.f, kv = (__bf16)0.f, vv = (__bf16)0.f;
  if (n < SEQ) {
    const __bf16* r = qkv + (size_t)(b * SEQ + n) * (3 * DIM) + h * HD + d;
    qv = r[0];
    kv = r[DIM];
    vv = r[2 * DIM];
  }
  qp[(size_t)(bh * NPADS + n) * HD + d] = qv;
  kp[(size_t)(bh * NPADS + n) * HD + d] = kv;
  vt[(size_t)bh * HD * NPADS + (size_t)d * NPADS + n] = vv;
}

// ---------------------------------------------------------------------------
// Attention: one block (128 thr / 4 waves) per (b,h, 16-query-row tile).
// Phase 1: S = scale * Q K^T (WMMA -> LDS fp32)
// Phase 2: split softmax ([0,20) and [20,596)); write fp32 weights + bf16 P
// Phase 3: O = P V (WMMA, A from LDS, B from V^T) -> bf16 o[BN,768]
// ---------------------------------------------------------------------------
__global__ __launch_bounds__(128) void attn_kernel(const __bf16* __restrict__ qp,
                                                   const __bf16* __restrict__ kp,
                                                   const __bf16* __restrict__ vt,
                                                   float* __restrict__ weights,
                                                   __bf16* __restrict__ o) {
  __shared__ __attribute__((aligned(16))) float  S[16 * LDP];
  __shared__ __attribute__((aligned(16))) __bf16 P[16 * LDP];

  const int qt   = blockIdx.x % (NPADS / 16);   // 38 query tiles
  const int bh   = blockIdx.x / (NPADS / 16);
  const int h    = bh % HEADS;
  const int b    = bh / HEADS;
  const int row0 = qt * 16;
  const int wid  = threadIdx.x >> 5;
  const int lane = threadIdx.x & 31;
  const float scale = 0.125f;                   // 64^-0.5

  const __bf16* qbase = qp + (size_t)(bh * NPADS + row0) * HD;
  const __bf16* kbase = kp + (size_t)bh * NPADS * HD;

  // ---- Phase 1: scores -> LDS
  for (int nt = wid; nt < NPADS / 16; nt += 4) {
    v8f c = {};
#pragma unroll
    for (int ks = 0; ks < HD; ks += 32) {
      v16bf a  = load_frag(qbase, HD, ks);
      v16bf bb = load_frag(kbase + (size_t)(nt * 16) * HD, HD, ks);
      c = wmma_bf16(a, bb, c);
    }
    const int n   = nt * 16 + (lane & 15);
    const int mhi = (lane >> 4) << 3;
#pragma unroll
    for (int r = 0; r < 8; ++r) S[(mhi + r) * LDP + n] = c[r] * scale;
  }
  __syncthreads();

  // ---- Phase 2: split softmax (one thread per query row)
  if (threadIdx.x < 16) {
    const int r    = threadIdx.x;
    const int qrow = row0 + r;
    float* Sr  = S + r * LDP;
    __bf16* Pr = P + r * LDP;
    if (qrow < SEQ) {
      // segment 1: cls columns [0, 20)
      float m1 = -3.4e38f;
      for (int c = 0; c < CLSK; ++c) m1 = fmaxf(m1, Sr[c]);
      float s1 = 0.f;
      for (int c = 0; c < CLSK; ++c) { float e = expf(Sr[c] - m1); Sr[c] = e; s1 += e; }
      const float i1 = 1.f / s1;
      // segment 2: patch columns [20, 596)
      float m2 = -3.4e38f;
      for (int c = CLSK; c < SEQ; ++c) m2 = fmaxf(m2, Sr[c]);
      float s2 = 0.f;
      for (int c = CLSK; c < SEQ; ++c) { float e = expf(Sr[c] - m2); Sr[c] = e; s2 += e; }
      const float i2 = 1.f / s2;
      float* wrow = weights + ((size_t)(b * HEADS + h) * SEQ + qrow) * SEQ;
      for (int c = 0; c < CLSK; ++c) { float p = Sr[c] * i1; wrow[c] = p; Pr[c] = (__bf16)p; }
      for (int c = CLSK; c < SEQ; ++c) { float p = Sr[c] * i2; wrow[c] = p; Pr[c] = (__bf16)p; }
      for (int c = SEQ; c < LDP; ++c) Pr[c] = (__bf16)0.f;
    } else {
      for (int c = 0; c < LDP; ++c) Pr[c] = (__bf16)0.f;
    }
  }
  __syncthreads();

  // ---- Phase 3: O = P @ V  (wave wid handles d-columns [wid*16, wid*16+16))
  v8f c = {};
  const __bf16* vbase = vt + (size_t)bh * HD * NPADS + (size_t)(wid * 16) * NPADS;
  for (int ks = 0; ks < NPADS; ks += 32) {
    v16bf a  = load_frag((const __bf16*)P, LDP, ks);
    v16bf bb = load_frag(vbase, NPADS, ks);
    c = wmma_bf16(a, bb, c);
  }
  const int d   = wid * 16 + (lane & 15);
  const int mhi = (lane >> 4) << 3;
#pragma unroll
  for (int r = 0; r < 8; ++r) {
    const int qrow = row0 + mhi + r;
    if (qrow < SEQ)
      o[(size_t)(b * SEQ + qrow) * DIM + h * HD + d] = (__bf16)c[r];
  }
}

// ---------------------------------------------------------------------------
// Host-side orchestration
// ---------------------------------------------------------------------------
extern "C" void kernel_launch(void* const* d_in, const int* in_sizes, int n_in,
                              void* d_out, int out_size, void* d_ws, size_t ws_size,
                              hipStream_t stream) {
  (void)in_sizes; (void)n_in; (void)out_size; (void)ws_size;

  const float* x      = (const float*)d_in[0];
  const float* qkv_w  = (const float*)d_in[1];
  const float* qkv_b  = (const float*)d_in[2];
  const float* proj_w = (const float*)d_in[3];
  const float* proj_b = (const float*)d_in[4];
  const float* ln1_g  = (const float*)d_in[5];
  const float* ln1_b  = (const float*)d_in[6];
  const float* ln2_g  = (const float*)d_in[7];
  const float* ln2_b  = (const float*)d_in[8];
  const float* fc1_w  = (const float*)d_in[9];
  const float* fc1_b  = (const float*)d_in[10];
  const float* fc2_w  = (const float*)d_in[11];
  const float* fc2_b  = (const float*)d_in[12];

  float* out_x = (float*)d_out;                         // [16,596,768]
  float* out_w = out_x + (size_t)BN * DIM;              // [16,12,596,596]

  // workspace carve-up
  size_t off = 0;
  auto carve = [&](size_t bytes) {
    void* p = (char*)d_ws + off;
    off += (bytes + 255) & ~(size_t)255;
    return p;
  };
  __bf16* qkv_w_bf = (__bf16*)carve((size_t)3 * DIM * DIM * 2);
  __bf16* proj_w_bf = (__bf16*)carve((size_t)DIM * DIM * 2);
  __bf16* fc1_w_bf = (__bf16*)carve((size_t)HIDDEN * DIM * 2);
  __bf16* fc2_w_bf = (__bf16*)carve((size_t)DIM * HIDDEN * 2);
  __bf16* xn   = (__bf16*)carve((size_t)BN * DIM * 2);
  __bf16* qkv  = (__bf16*)carve((size_t)BN * 3 * DIM * 2);
  __bf16* qp   = (__bf16*)carve((size_t)BATCH * HEADS * NPADS * HD * 2);
  __bf16* kp   = (__bf16*)carve((size_t)BATCH * HEADS * NPADS * HD * 2);
  __bf16* vt   = (__bf16*)carve((size_t)BATCH * HEADS * NPADS * HD * 2);
  __bf16* obuf = (__bf16*)carve((size_t)BN * DIM * 2);
  float*  x1   = (float*)carve((size_t)BN * DIM * 4);
  __bf16* hbuf = (__bf16*)carve((size_t)BN * DIM * 2);
  __bf16* h1   = (__bf16*)carve((size_t)BN * HIDDEN * 2);

  // 1) weights fp32 -> bf16
  auto cvt = [&](const float* src, __bf16* dst, int n) {
    cvt_bf16_kernel<<<(n + 255) / 256, 256, 0, stream>>>(src, dst, n);
  };
  cvt(qkv_w,  qkv_w_bf,  3 * DIM * DIM);
  cvt(proj_w, proj_w_bf, DIM * DIM);
  cvt(fc1_w,  fc1_w_bf,  HIDDEN * DIM);
  cvt(fc2_w,  fc2_w_bf,  DIM * HIDDEN);

  // 2) LN1
  ln_kernel<<<BN, 256, 0, stream>>>(x, ln1_g, ln1_b, xn);

  // 3) QKV GEMM: [9536,2304] = xn @ qkv_w^T + b   (bf16 out)
  gemm_wmma_kernel<<<dim3((3 * DIM) / 128, BN / 64), 128, 0, stream>>>(
      xn, qkv_w_bf, qkv_b, nullptr, nullptr, qkv, BN, 3 * DIM, DIM, 0);

  // 4) repack into padded Q/K and transposed V
  {
    const int tot = BATCH * HEADS * NPADS * HD;
    repack_kernel<<<(tot + 255) / 256, 256, 0, stream>>>(qkv, qp, kp, vt);
  }

  // 5) attention (scores + split softmax + P@V), writes weights fp32
  attn_kernel<<<BATCH * HEADS * (NPADS / 16), 128, 0, stream>>>(
      qp, kp, vt, out_w, obuf);

  // 6) proj GEMM + residual x -> x1 (fp32)
  gemm_wmma_kernel<<<dim3(DIM / 128, BN / 64), 128, 0, stream>>>(
      obuf, proj_w_bf, proj_b, x, x1, nullptr, BN, DIM, DIM, 0);

  // 7) LN2 on x1 -> hbuf (bf16)
  ln_kernel<<<BN, 256, 0, stream>>>(x1, ln2_g, ln2_b, hbuf);

  // 8) FC1 + exact GELU -> h1 (bf16)
  gemm_wmma_kernel<<<dim3(HIDDEN / 128, BN / 64), 128, 0, stream>>>(
      hbuf, fc1_w_bf, fc1_b, nullptr, nullptr, h1, BN, HIDDEN, DIM, 1);

  // 9) FC2 + residual x1 -> d_out (fp32)
  gemm_wmma_kernel<<<dim3(DIM / 128, BN / 64), 128, 0, stream>>>(
      h1, fc2_w_bf, fc2_b, x1, out_x, nullptr, BN, DIM, HIDDEN, 0);
}
